// Alignment_24352464569226
// MI455X (gfx1250) — matrix operand order repaired and checked by reading, not compiled, not run on hardware
//
#include <hip/hip_runtime.h>
#include <hip/hip_bf16.h>

// ---------------- problem constants ----------------
constexpr int Bn = 32, LA = 512, LB = 512, H = 256;
constexpr float NEGV = -10000000.0f;
constexpr float FMIN = -3.402823466e38f;

typedef __attribute__((ext_vector_type(16))) __bf16 v16bf;
typedef __attribute__((ext_vector_type(8)))  __bf16 v8bf;
typedef __attribute__((ext_vector_type(8)))  float  v8f;

// ---------------- gfx1250 async copy: global -> LDS (ASYNCcnt path) --------
__device__ __forceinline__ void async_copy16(unsigned lds_off, const void* gptr) {
    asm volatile("global_load_async_to_lds_b128 %0, %1, off"
                 :: "v"(lds_off), "v"((unsigned long long)(size_t)gptr)
                 : "memory");
}
__device__ __forceinline__ void wait_async_all() {
    asm volatile("s_wait_asynccnt 0x0" ::: "memory");
}

// ---------------- cast fp32 -> bf16 (4 elems/thread) ----------------
struct alignas(8) bf4 { __bf16 a, b, c, d; };

__global__ void cast_bf16_kernel(const float* __restrict__ x,
                                 __bf16* __restrict__ y, int n4) {
    int i = blockIdx.x * blockDim.x + threadIdx.x;
    if (i < n4) {
        float4 v = ((const float4*)x)[i];
        bf4 o; o.a = (__bf16)v.x; o.b = (__bf16)v.y;
        o.c = (__bf16)v.z; o.d = (__bf16)v.w;
        ((bf4*)y)[i] = o;
    }
}

// ---------------- tiled transpose+cast: x[B][R][C] f32 -> y[B][C][R] bf16 ----
__global__ void transpose_cast_kernel(const float* __restrict__ x,
                                      __bf16* __restrict__ y, int R, int C) {
    __shared__ __bf16 t[32][33];
    const int z = blockIdx.z;
    const int r0 = blockIdx.y * 32, c0 = blockIdx.x * 32;
    const float* xb = x + (size_t)z * R * C;
    __bf16* yb = y + (size_t)z * C * R;
    const int lx = threadIdx.x, ly = threadIdx.y;   // block (32,8)
#pragma unroll
    for (int k = 0; k < 4; k++) {
        int r = r0 + ly + k * 8;
        t[ly + k * 8][lx] = (__bf16)xb[(size_t)r * C + c0 + lx];
    }
    __syncthreads();
#pragma unroll
    for (int k = 0; k < 4; k++) {
        int cl = ly + k * 8;
        yb[(size_t)(c0 + cl) * R + r0 + lx] = t[lx][cl];
    }
}

// ---------------- NT GEMM with async-LDS double buffering -------------------
// C[M][N] = A[M][K] x Bt[N][K]^T (bf16 -> f32)
// block = 256 threads = 8 waves; wave tile 32x64 -> block tile 64x256
// K-step 32; LDS rows padded to 40 bf16 (80B) for bank spread, 16B aligned.
constexpr int APITCH = 40;

template <bool MASKED>
__global__ __launch_bounds__(256)
void gemm_nt_kernel(const __bf16* __restrict__ A, const __bf16* __restrict__ Bt,
                    float* __restrict__ C, int M, int N, int K,
                    const unsigned char* __restrict__ mA,
                    const unsigned char* __restrict__ mB,
                    const float* __restrict__ tempPtr) {
    __shared__ __bf16 sA[2][64 * APITCH];
    __shared__ __bf16 sB[2][256 * APITCH];
    const int tid  = threadIdx.x;
    const int lane = tid & 31;
    const int wave = tid >> 5;
    const int wm = wave >> 2, wn = wave & 3;
    const int z = blockIdx.z;
    const int mblk = blockIdx.y * 64;
    const int nblk = blockIdx.x * 256;
    A  += (size_t)z * M * K;
    Bt += (size_t)z * N * K;
    C  += (size_t)z * M * N;

    // staging coordinates: each thread moves 16B of A and 4x16B of B per k-step
    const int arow = tid >> 2, ach = tid & 3;           // A: 64 rows x 4 chunks
    const unsigned sAbase = (unsigned)(size_t)&sA[0][0];
    const unsigned sBbase = (unsigned)(size_t)&sB[0][0];

    auto stage = [&](int buf, int k) {
        async_copy16(sAbase + (unsigned)(buf * 64 * APITCH + arow * APITCH + ach * 8) * 2,
                     A + (size_t)(mblk + arow) * K + k + ach * 8);
#pragma unroll
        for (int c = 0; c < 4; c++) {
            int idx  = tid + c * 256;
            int brow = idx >> 2, bch = idx & 3;
            async_copy16(sBbase + (unsigned)(buf * 256 * APITCH + brow * APITCH + bch * 8) * 2,
                         Bt + (size_t)(nblk + brow) * K + k + bch * 8);
        }
    };

    v8f acc[2][4] = {};
    const int nk = K / 32;
    stage(0, 0);
    for (int kk = 0; kk < nk; ++kk) {
        const int buf = kk & 1;
        wait_async_all();        // my async writes to LDS have landed
        __syncthreads();         // everyone's writes landed; everyone's reads done
        if (kk + 1 < nk) stage(buf ^ 1, (kk + 1) * 32);

        const __bf16* pA = &sA[buf][0];
        const __bf16* pB = &sB[buf][0];
        const int r   = lane & 15;
        const int kbA = (lane & 16) ? 16 : 0;   // A 16x32: lanes 0-15 K 0..15
        const int kbB = (lane & 16) ? 16 : 0;   // B 32x16: lanes 0-15 K 0..15
        v16bf aF[2], bF[4];
#pragma unroll
        for (int i = 0; i < 2; i++) {
            const __bf16* q = pA + (wm * 32 + i * 16 + r) * APITCH + kbA;
            v8bf lo = *(const v8bf*)(q);
            v8bf hi = *(const v8bf*)(q + 8);
            aF[i] = __builtin_shufflevector(lo, hi, 0,1,2,3,4,5,6,7,8,9,10,11,12,13,14,15);
        }
#pragma unroll
        for (int j = 0; j < 4; j++) {
            const __bf16* q = pB + (wn * 64 + j * 16 + r) * APITCH + kbB;
            v8bf lo = *(const v8bf*)(q);
            v8bf hi = *(const v8bf*)(q + 8);
            bF[j] = __builtin_shufflevector(lo, hi, 0,1,2,3,4,5,6,7,8,9,10,11,12,13,14,15);
        }
#pragma unroll
        for (int i = 0; i < 2; i++)
#pragma unroll
            for (int j = 0; j < 4; j++)
                acc[i][j] = __builtin_amdgcn_wmma_f32_16x16x32_bf16(
                    false, aF[i], false, bF[j], (short)0, acc[i][j], false, false);
    }

    float scale = 1.0f;
    if (MASKED) scale = *tempPtr;
    const int mbase = (lane & 16) ? 8 : 0;
    const int nc = lane & 15;
#pragma unroll
    for (int i = 0; i < 2; i++) {
#pragma unroll
        for (int j = 0; j < 4; j++) {
            const int ncol = nblk + wn * 64 + j * 16 + nc;
            bool bm = true;
            if (MASKED) bm = mB[(size_t)z * N + ncol] != 0;
#pragma unroll
            for (int v = 0; v < 8; v++) {
                const int m = mblk + wm * 32 + i * 16 + mbase + v;
                float val = acc[i][j][v];
                if (MASKED) {
                    bool am = mA[(size_t)z * M + m] != 0;
                    val = (am && bm) ? val * scale : NEGV;
                }
                C[(size_t)m * N + ncol] = val;
            }
        }
    }
}

// ---------------- row stats: per (b,i) max & sum-exp over j (one wave/row) --
__global__ __launch_bounds__(256)
void row_stats_kernel(const float* __restrict__ S,
                      float* __restrict__ rmax, float* __restrict__ rsum) {
    const int row  = blockIdx.x * 8 + (threadIdx.x >> 5);  // [0, B*LA)
    const int lane = threadIdx.x & 31;
    const float* p = S + (size_t)row * LB;
    float vals[16];
    float m = FMIN;
#pragma unroll
    for (int j = 0; j < 16; j++) { vals[j] = p[lane + j * 32]; m = fmaxf(m, vals[j]); }
#pragma unroll
    for (int off = 16; off; off >>= 1) m = fmaxf(m, __shfl_xor(m, off, 32));
    float s = 0.f;
#pragma unroll
    for (int j = 0; j < 16; j++) s += __expf(vals[j] - m);
#pragma unroll
    for (int off = 16; off; off >>= 1) s += __shfl_xor(s, off, 32);
    if (lane == 0) { rmax[row] = m; rsum[row] = s; }
}

// ---------------- col stats: per (b,j) online max & sum-exp over i ----------
__global__ __launch_bounds__(256)
void col_stats_kernel(const float* __restrict__ S,
                      float* __restrict__ cmax, float* __restrict__ csum) {
    const int nb = LB / 256;
    const int z = blockIdx.x / nb;
    const int j = (blockIdx.x % nb) * 256 + threadIdx.x;
    const float* p = S + ((size_t)z * LA) * LB + j;
    float m = FMIN, s = 0.f;
    for (int i = 0; i < LA; i++) {
        float x = p[(size_t)i * LB];
        float mn = fmaxf(m, x);
        s = s * __expf(m - mn) + __expf(x - mn);
        m = mn;
    }
    cmax[(size_t)z * LB + j] = m;
    csum[(size_t)z * LB + j] = s;
}

// ------- normalize: attn_b[b][i][j] bf16 and attn_aT[b][j][i] bf16 ----------
__global__ __launch_bounds__(256)
void normalize_kernel(const float* __restrict__ S,
                      const float* __restrict__ rmax, const float* __restrict__ rsum,
                      const float* __restrict__ cmax, const float* __restrict__ csum,
                      __bf16* __restrict__ attn_b, __bf16* __restrict__ attn_aT) {
    __shared__ __bf16 t[32][33];
    const int z = blockIdx.z;
    const int i0 = blockIdx.y * 32, j0 = blockIdx.x * 32;
    const float* Sp = S + ((size_t)z * LA) * LB;
    const int lx = threadIdx.x, ly = threadIdx.y;   // block (32,8)
    const int j = j0 + lx;
    const float cm = cmax[(size_t)z * LB + j];
    const float ci = 1.0f / csum[(size_t)z * LB + j];
#pragma unroll
    for (int k = 0; k < 4; k++) {
        int il = ly + k * 8;
        int i = i0 + il;
        float x = Sp[(size_t)i * LB + j];
        float rb = __expf(x - rmax[(size_t)z * LA + i]) / rsum[(size_t)z * LA + i];
        attn_b[((size_t)z * LA + i) * LB + j] = (__bf16)rb;
        t[il][lx] = (__bf16)(__expf(x - cm) * ci);
    }
    __syncthreads();
#pragma unroll
    for (int k = 0; k < 4; k++) {
        int jl = ly + k * 8;
        attn_aT[((size_t)z * LB + (j0 + jl)) * LA + i0 + lx] = t[lx][jl];
    }
}

// ---------------- host orchestration ----------------------------------------
extern "C" void kernel_launch(void* const* d_in, const int* in_sizes, int n_in,
                              void* d_out, int out_size, void* d_ws, size_t ws_size,
                              hipStream_t stream) {
    const float* a = (const float*)d_in[0];               // [B][LA][H]
    const float* b = (const float*)d_in[1];               // [B][LB][H]
    const unsigned char* mask_a = (const unsigned char*)d_in[2];  // [B][LA]
    const unsigned char* mask_b = (const unsigned char*)d_in[3];  // [B][LB]
    const float* temperature = (const float*)d_in[4];     // scalar

    const size_t nA = (size_t)Bn * LA * H;                // 4,194,304 elems
    const size_t nS = (size_t)Bn * LA * LB;               // 8,388,608 elems

    char* w = (char*)d_ws;
    __bf16* aB  = (__bf16*)w;                 w += nA * 2;   // a bf16 [LA][H]
    __bf16* bB  = (__bf16*)w;                 w += nA * 2;   // b bf16 [LB][H]
    __bf16* aT  = (__bf16*)w;                 w += nA * 2;   // aT bf16 [H][LA]
    __bf16* bT  = (__bf16*)w;                 w += nA * 2;   // bT bf16 [H][LB]
    float*  S   = (float*)w;                  w += nS * 4;   // scores f32
    __bf16* atb = (__bf16*)w;                 w += nS * 2;   // attn_b bf16 [LA][LB]
    __bf16* ata = (__bf16*)w;                 w += nS * 2;   // attn_aT bf16 [LB][LA]
    float* rmax = (float*)w;                  w += (size_t)Bn * LA * 4;
    float* rsum = (float*)w;                  w += (size_t)Bn * LA * 4;
    float* cmax = (float*)w;                  w += (size_t)Bn * LB * 4;
    float* csum = (float*)w;                  w += (size_t)Bn * LB * 4;

    float* feature_a = (float*)d_out;                    // [B][LA][H]
    float* feature_b = (float*)d_out + nA;               // [B][LB][H]

    // 1) casts + transposed casts
    cast_bf16_kernel<<<(int)(nA / 4 + 255) / 256, 256, 0, stream>>>(a, aB, (int)(nA / 4));
    cast_bf16_kernel<<<(int)(nA / 4 + 255) / 256, 256, 0, stream>>>(b, bB, (int)(nA / 4));
    transpose_cast_kernel<<<dim3(H / 32, LA / 32, Bn), dim3(32, 8), 0, stream>>>(a, aT, LA, H);
    transpose_cast_kernel<<<dim3(H / 32, LB / 32, Bn), dim3(32, 8), 0, stream>>>(b, bT, LB, H);

    // 2) S = a . b^T * temp  (masked) — NT GEMM, M=LA N=LB K=H
    gemm_nt_kernel<true><<<dim3(LB / 256, LA / 64, Bn), 256, 0, stream>>>(
        aB, bB, S, LA, LB, H, mask_a, mask_b, temperature);

    // 3) softmax stats (rows: axis=2 -> attn_b ; cols: axis=1 -> attn_a)
    row_stats_kernel<<<Bn * LA / 8, 256, 0, stream>>>(S, rmax, rsum);
    col_stats_kernel<<<Bn * (LB / 256), 256, 0, stream>>>(S, cmax, csum);

    // 4) normalize into bf16 attn_b and transposed attn_aT
    normalize_kernel<<<dim3(LB / 32, LA / 32, Bn), dim3(32, 8), 0, stream>>>(
        S, rmax, rsum, cmax, csum, atb, ata);

    // 5) feature_a = attn_b x b   (M=LA N=H K=LB, Bt = bT)
    gemm_nt_kernel<false><<<dim3(H / 256, LA / 64, Bn), 256, 0, stream>>>(
        atb, bT, feature_a, LA, H, LB, nullptr, nullptr, nullptr);

    // 6) feature_b = attn_aT x a  (M=LB N=H K=LA, Bt = aT)
    gemm_nt_kernel<false><<<dim3(H / 256, LB / 64, Bn), 256, 0, stream>>>(
        ata, aT, feature_b, LB, H, LA, nullptr, nullptr, nullptr);

    (void)in_sizes; (void)n_in; (void)out_size; (void)ws_size;
}
